// AttnDecoderRNN_13743895347513
// MI455X (gfx1250) — matrix operand order repaired and checked by reading, not compiled, not run on hardware
//
#include <hip/hip_runtime.h>
#include <math.h>

#define BSZ 32
#define HSZ 256
#define NVOCAB 128
#define HD 16
#define WD 64
#define ENC 1024
#define SPT (HD*WD)   // 1024 spatial positions
#define LDA 40        // LDS row pitch in bf16 elements (80B = 16B-aligned rows)

typedef __bf16 bf16;
typedef __attribute__((ext_vector_type(16))) __bf16 v16bf;
typedef __attribute__((ext_vector_type(8)))  float  v8f;

union Frag16 { v16bf v; unsigned int u[8]; };
union FragC  { v8f   v; float f[8]; };

// ---------- bf16 <-> f32 (bit-exact RNE, no reliance on __bf16 arithmetic) ----------
__device__ inline bf16 f2bf(float f) {
    unsigned u = __float_as_uint(f);
    unsigned r = (u + 0x7FFFu + ((u >> 16) & 1u)) >> 16;
    unsigned short us = (unsigned short)r;
    return *(bf16*)&us;
}
__device__ inline float bf2f(bf16 h) {
    unsigned short us = *(unsigned short*)&h;
    return __uint_as_float(((unsigned)us) << 16);
}

// ---------- WMMA fragment loaders (CDNA5 16-bit layouts, wave32) ----------
// A 16x32: lanes 0-15 M=lane, K in {0..7,16..23}; lanes 16-31 M=lane-16, K in {8..15,24..31}
__device__ inline void load_a_frag(Frag16& a, const bf16* tile, int lda, int m0, int lane) {
    int half = lane >> 4;
    const bf16* row = tile + (m0 + (lane & 15)) * lda;
#pragma unroll
    for (int vv = 0; vv < 4; ++vv)
        a.u[vv] = *(const unsigned int*)(row + (vv * 2 + half * 8));
#pragma unroll
    for (int vv = 4; vv < 8; ++vv)
        a.u[vv] = *(const unsigned int*)(row + ((vv - 4) * 2 + 16 + half * 8));
}
// B 32x16: N = lane%16; element s: K = 8*half+s (s<8), 16+8*half+(s-8) (s>=8). Bg points at (k0,n0), ldb = row stride.
__device__ inline void load_b_frag(Frag16& b, const bf16* Bg, int ldb, int lane) {
    int half = lane >> 4;
    const bf16* col = Bg + (lane & 15);
#pragma unroll
    for (int s = 0; s < 16; ++s)
        b.v[s] = col[(s < 8 ? 8 * half + s : 16 + 8 * half + (s - 8)) * ldb];
}

// ---------- small kernels ----------
__global__ void k_init(float* bnsum, float* bnsq, float* esum) {
    int i = threadIdx.x;
    if (i < 256) { bnsum[i] = 0.f; bnsq[i] = 0.f; }
    if (i < BSZ) esum[i] = 0.f;
}

__global__ void k_prep(const int* input_a, const float* embedding,
                       const float* attsum_in, const float* dec_att,
                       const float* c1w, const float* c1b,
                       const int* h_mask, const int* w_mask,
                       float* emb, float* mask, float* attsum_out) {
    int b = blockIdx.x, tid = threadIdx.x;
    if (tid < 256) emb[b * 256 + tid] = embedding[input_a[b] * 256 + tid];
    int hm = h_mask[b], wm = w_mask[b];
    for (int m = tid; m < SPT; m += blockDim.x) {
        int hh = m >> 6, ww = m & 63;
        mask[b * SPT + m] = (hh < hm && ww < wm) ? 1.f : 0.f;
        float s = c1b[0];
        for (int dy = -1; dy <= 1; ++dy)
            for (int dx = -1; dx <= 1; ++dx) {
                int y = hh + dy, x = ww + dx;
                if (y >= 0 && y < HD && x >= 0 && x < WD)
                    s += dec_att[b * SPT + y * WD + x] * c1w[(dy + 1) * 3 + (dx + 1)];
            }
        attsum_out[b * SPT + m] = attsum_in[b * SPT + m] + s;
    }
}

// weight conversion: ua.w (1024x256) -> bf16; conv_tan.w OIHW -> [tap][ci][co] bf16
__global__ void k_cvt(const float* uaw, const float* wtan, bf16* uaw_bf, bf16* wtan_bf) {
    int i = blockIdx.x * blockDim.x + threadIdx.x;
    if (i < ENC * 256) uaw_bf[i] = f2bf(uaw[i]);
    if (i < 256 * 256 * 9) {
        int co = i / (256 * 9);
        int rem = i % (256 * 9);
        int ci = rem / 9;
        int t = rem % 9;
        wtan_bf[(t * 256 + ci) * 256 + co] = f2bf(wtan[i]);
    }
}

// generic GRU-with-gate-memory cell (one thread per (b,i))
__global__ void k_gru(const float* x, int K, const float* hprev, const float* vprev,
                      const float* wi, const float* wh, const float* wv,
                      const float* bi, const float* bh, float* hnew, float* vnew) {
    int idx = blockIdx.x * blockDim.x + threadIdx.x;
    if (idx >= BSZ * HSZ) return;
    int b = idx / HSZ, i = idx % HSZ;
    const float* xb = x + (size_t)b * K;
    const float* hb = hprev + b * HSZ;
    const float* vb = vprev + b * 3 * HSZ;
    float sR = bi[i] + bh[i];
    float sZ = bi[i + HSZ] + bh[i + HSZ];
    float giN = bi[i + 2 * HSZ];
    float ghN = bh[i + 2 * HSZ];
    float gvN = 0.f;
    for (int k = 0; k < K; ++k) {
        float xv = xb[k];
        const float* wr = wi + (size_t)k * 768;
        sR += xv * wr[i]; sZ += xv * wr[i + 256]; giN += xv * wr[i + 512];
    }
    for (int k = 0; k < HSZ; ++k) {
        float hv = hb[k];
        const float* wr = wh + (size_t)k * 768;
        sR += hv * wr[i]; sZ += hv * wr[i + 256]; ghN += hv * wr[i + 512];
    }
    for (int k = 0; k < 768; ++k) {
        float vv = vb[k];
        const float* wr = wv + (size_t)k * 768;
        sR += vv * wr[i]; sZ += vv * wr[i + 256]; gvN += vv * wr[i + 512];
    }
    float r = 1.f / (1.f + expf(-sR));
    float z = 1.f / (1.f + expf(-sZ));
    float n = tanhf(giN + r * ghN + gvN);
    hnew[b * HSZ + i] = (1.f - z) * n + z * hb[i];
    vnew[b * 768 + i] = r;
    vnew[b * 768 + 256 + i] = z;
    vnew[b * 768 + 512 + i] = n;
}

__global__ void k_hidden1(const float* st, const float* w, const float* bias, float* out) {
    int idx = blockIdx.x * blockDim.x + threadIdx.x;
    if (idx >= BSZ * 256) return;
    int b = idx / 256, i = idx % 256;
    float s = bias[i];
    const float* sb = st + b * 256;
    for (int k = 0; k < 256; ++k) s += sb[k] * w[k * 256 + i];
    out[idx] = s;
}

// ---------- WMMA kernel 1: et = hidden1 + enc_t@ua.w + ua.b + att*uf.w + uf.b ----------
__global__ __launch_bounds__(256) void k_ua(const float* enc, const bf16* uaw_bf,
                                            const float* uab, const float* ufw, const float* ufb,
                                            const float* hidden1, const float* attsum, bf16* et) {
    __shared__ __align__(16) bf16 Atile[64 * LDA];
    int blk = blockIdx.x;
    int b = blk >> 4, h = blk & 15;
    int tid = threadIdx.x, lane = tid & 31, wave = tid >> 5;
    int m_sub = wave & 3, n_half = wave >> 2;
    FragC acc[8];
    for (int j = 0; j < 8; ++j)
        for (int r = 0; r < 8; ++r) acc[j].f[r] = 0.f;
    const float* encb = enc + (size_t)b * ENC * SPT + h * WD;   // enc[b][k][h*64 + w]
    for (int ks = 0; ks < ENC / 32; ++ks) {
        __syncthreads();
        {   // stage A: transpose k-major global -> [m][k] LDS, fp32 -> bf16
            int kk = tid >> 3;
            int m8 = (tid & 7) * 8;
            const float* src = encb + (size_t)(ks * 32 + kk) * SPT + m8;
#pragma unroll
            for (int j = 0; j < 8; ++j) Atile[(m8 + j) * LDA + kk] = f2bf(src[j]);
        }
        __syncthreads();
        Frag16 af;
        load_a_frag(af, Atile, LDA, m_sub * 16, lane);
        const bf16* Bg = uaw_bf + (size_t)(ks * 32) * 256;
#pragma unroll
        for (int j = 0; j < 8; ++j) {
            int n0 = (n_half * 8 + j) * 16;
            Frag16 bfr;
            load_b_frag(bfr, Bg + n0, 256, lane);
            acc[j].v = __builtin_amdgcn_wmma_f32_16x16x32_bf16(
                false, af.v, false, bfr.v, (short)0, acc[j].v, false, false);
        }
    }
    int half = lane >> 4, ncol = lane & 15;
    for (int j = 0; j < 8; ++j) {
        int N = (n_half * 8 + j) * 16 + ncol;
        float base = hidden1[b * 256 + N] + uab[N] + ufb[N];
        float ufwN = ufw[N];
#pragma unroll
        for (int r = 0; r < 8; ++r) {
            int M = m_sub * 16 + r + half * 8;   // w coordinate
            float attv = attsum[b * SPT + h * WD + M];
            float val = acc[j].f[r] + base + attv * ufwN;
            et[(((size_t)b * HD + h) * WD + M) * 256 + N] = f2bf(val);
        }
    }
}

// ---------- WMMA kernel 2: implicit-GEMM 3x3 conv (256->256) + bias, * mask, BN sums ----
// A-staging uses the CDNA5 async global->LDS engine (ASYNCcnt) for interior chunks.
__global__ __launch_bounds__(256) void k_conv(const bf16* et, const bf16* wt_bf,
                                              const float* convb, const float* mask,
                                              bf16* xconv, float* bnsum, float* bnsq) {
    __shared__ __align__(16) bf16 Atile[64 * LDA];
    int blk = blockIdx.x;
    int b = blk >> 4, h = blk & 15;
    int tid = threadIdx.x, lane = tid & 31, wave = tid >> 5;
    int m_sub = wave & 3, n_half = wave >> 2;
    FragC acc[8];
    for (int j = 0; j < 8; ++j)
        for (int r = 0; r < 8; ++r) acc[j].f[r] = 0.f;
    // precompute this thread's staging slot: row m (0..63), 16B chunk dstart (dword units)
    int sm = tid >> 2;
    int dstart = (tid & 3) * 4;                 // dword index within the 64B row chunk
    unsigned lds_dst = (unsigned)(size_t)(Atile + sm * LDA) + (unsigned)(dstart * 4);
    for (int t = 0; t < 9; ++t) {
        int dy = t / 3 - 1, dx = t % 3 - 1;
        int hh = h + dy;
        int wsrc = sm + dx;
        bool ok = (hh >= 0 && hh < HD && wsrc >= 0 && wsrc < WD);
        const bf16* rowsrc = et + (((size_t)b * HD + hh) * WD + wsrc) * 256;
        for (int ks = 0; ks < 256 / 32; ++ks) {
            __syncthreads();
            if (ok) {
                // async copy 16B global -> LDS (bf16 passthrough, no VGPR round-trip)
                unsigned long long gsrc =
                    (unsigned long long)(size_t)(rowsrc + ks * 32 + dstart * 2);
                asm volatile("global_load_async_to_lds_b128 %0, %1, off"
                             :: "v"(lds_dst), "v"(gsrc)
                             : "memory");
            } else {
                unsigned int* dst = (unsigned int*)(Atile + sm * LDA);
#pragma unroll
                for (int d = 0; d < 4; ++d) dst[dstart + d] = 0u;
            }
            asm volatile("s_wait_asynccnt 0x0" ::: "memory");
            __syncthreads();
            Frag16 af;
            load_a_frag(af, Atile, LDA, m_sub * 16, lane);
            const bf16* Bg = wt_bf + ((size_t)t * 256 + ks * 32) * 256;
#pragma unroll
            for (int j = 0; j < 8; ++j) {
                int n0 = (n_half * 8 + j) * 16;
                Frag16 bfr;
                load_b_frag(bfr, Bg + n0, 256, lane);
                acc[j].v = __builtin_amdgcn_wmma_f32_16x16x32_bf16(
                    false, af.v, false, bfr.v, (short)0, acc[j].v, false, false);
            }
        }
    }
    int half = lane >> 4, ncol = lane & 15;
    for (int j = 0; j < 8; ++j) {
        int N = (n_half * 8 + j) * 16 + ncol;
        float bias = convb[N];
        float s = 0.f, sq = 0.f;
#pragma unroll
        for (int r = 0; r < 8; ++r) {
            int M = m_sub * 16 + r + half * 8;
            float mval = mask[b * SPT + h * WD + M];
            float val = (acc[j].f[r] + bias) * mval;
            xconv[(((size_t)b * HD + h) * WD + M) * 256 + N] = f2bf(val);
            s += val;
            sq += val * val;
        }
        atomicAdd(&bnsum[N], s);
        atomicAdd(&bnsq[N], sq);
    }
}

__global__ void k_bnfin(const float* bnsum, const float* bnsq, float* mu, float* rstd) {
    int c = threadIdx.x;
    if (c >= 256) return;
    const float cnt = (float)(BSZ * SPT);
    float m = bnsum[c] / cnt;
    float v = bnsq[c] / cnt - m * m;
    mu[c] = m;
    rstd[c] = rsqrtf(v + 1e-5f);
}

__global__ void k_score(const bf16* xconv, const float* mu, const float* rstd,
                        const float* gamma, const float* beta,
                        const float* vattw, const float* vattb, const float* mask,
                        float* e_ws, float* esum) {
    int idx = blockIdx.x * blockDim.x + threadIdx.x;
    if (idx >= BSZ * SPT) return;
    const bf16* row = xconv + (size_t)idx * 256;
    float s = vattb[0];
    for (int c = 0; c < 256; ++c) {
        float x = bf2f(row[c]);
        float xn = (x - mu[c]) * rstd[c] * gamma[c] + beta[c];
        s += tanhf(xn) * vattw[c];
    }
    int b = idx >> 10;
    float e = expf(s) * mask[idx];
    e_ws[idx] = e;
    atomicAdd(&esum[b], e);
}

__global__ void k_alpha(const float* e_ws, const float* esum, float* alpha) {
    int idx = blockIdx.x * blockDim.x + threadIdx.x;
    if (idx >= BSZ * SPT) return;
    alpha[idx] = e_ws[idx] / (esum[idx >> 10] + 1e-8f);
}

__global__ void k_ct(const float* alpha, const float* enc, float* ct) {
    int idx = blockIdx.x * blockDim.x + threadIdx.x;
    if (idx >= BSZ * ENC) return;
    int b = idx / ENC, c = idx % ENC;
    const float* ap = alpha + b * SPT;
    const float* ep = enc + ((size_t)b * ENC + c) * SPT;
    float s = 0.f;
    for (int m = 0; m < SPT; m += 128) {
        __builtin_prefetch(ep + m + 128, 0, 0);
#pragma unroll 4
        for (int k = 0; k < 128; ++k) s += ap[m + k] * ep[m + k];
    }
    ct[idx] = s;
}

__global__ void k_prelogit(const float* h1, const float* emb, const float* ct,
                           const float* h2w, const float* h2b,
                           const float* e2w, const float* e2b,
                           const float* wcw, const float* wcb, float* pre) {
    int idx = blockIdx.x * blockDim.x + threadIdx.x;
    if (idx >= BSZ * 128) return;
    int b = idx / 128, j = idx % 128;
    float s = h2b[j] + e2b[j] + wcb[j];
    for (int i = 0; i < 256; ++i)
        s += h1[b * 256 + i] * h2w[i * 128 + j] + emb[b * 256 + i] * e2w[i * 128 + j];
    for (int k = 0; k < 1024; ++k) s += ct[b * 1024 + k] * wcw[k * 128 + j];
    pre[idx] = s;
}

__global__ void k_out(const float* pre, const float* outw, const float* outb, float* out) {
    __shared__ float red[128];
    int b = blockIdx.x, k = threadIdx.x;
    float lg = outb[k];
    for (int j = 0; j < 128; ++j) lg += pre[b * 128 + j] * outw[j * 128 + k];
    red[k] = lg;
    __syncthreads();
    for (int s = 64; s > 0; s >>= 1) {
        if (k < s) red[k] = fmaxf(red[k], red[k + s]);
        __syncthreads();
    }
    float mx = red[0];
    __syncthreads();
    red[k] = expf(lg - mx);
    __syncthreads();
    for (int s = 64; s > 0; s >>= 1) {
        if (k < s) red[k] += red[k + s];
        __syncthreads();
    }
    float lse = mx + logf(red[0]);
    out[b * 128 + k] = lg - lse;
}

extern "C" void kernel_launch(void* const* d_in, const int* in_sizes, int n_in,
                              void* d_out, int out_size, void* d_ws, size_t ws_size,
                              hipStream_t stream) {
    (void)in_sizes; (void)n_in; (void)out_size; (void)ws_size;
    // params (setup_inputs insertion order)
    const float* P_embedding = (const float*)d_in[0];
    const float* P_gru_wi  = (const float*)d_in[1];
    const float* P_gru_wh  = (const float*)d_in[2];
    const float* P_gru_wv  = (const float*)d_in[3];
    const float* P_gru_bi  = (const float*)d_in[4];
    const float* P_gru_bh  = (const float*)d_in[5];
    const float* P_gru1_wi = (const float*)d_in[6];
    const float* P_gru1_wh = (const float*)d_in[7];
    const float* P_gru1_wv = (const float*)d_in[8];
    const float* P_gru1_bi = (const float*)d_in[9];
    const float* P_gru1_bh = (const float*)d_in[10];
    const float* P_out_w   = (const float*)d_in[11];
    const float* P_out_b   = (const float*)d_in[12];
    const float* P_hid_w   = (const float*)d_in[13];
    const float* P_hid_b   = (const float*)d_in[14];
    const float* P_c1_w    = (const float*)d_in[15];
    const float* P_c1_b    = (const float*)d_in[16];
    const float* P_ctan_w  = (const float*)d_in[17];
    const float* P_ctan_b  = (const float*)d_in[18];
    const float* P_h2_w    = (const float*)d_in[19];
    const float* P_h2_b    = (const float*)d_in[20];
    const float* P_e2_w    = (const float*)d_in[21];
    const float* P_e2_b    = (const float*)d_in[22];
    const float* P_ua_w    = (const float*)d_in[23];
    const float* P_ua_b    = (const float*)d_in[24];
    const float* P_uf_w    = (const float*)d_in[25];
    const float* P_uf_b    = (const float*)d_in[26];
    const float* P_vatt_w  = (const float*)d_in[27];
    const float* P_vatt_b  = (const float*)d_in[28];
    const float* P_wc_w    = (const float*)d_in[29];
    const float* P_wc_b    = (const float*)d_in[30];
    const float* P_gamma   = (const float*)d_in[31];
    const float* P_beta    = (const float*)d_in[32];
    // runtime inputs
    const int*   I_input_a = (const int*)d_in[33];
    const float* I_hidden  = (const float*)d_in[34];
    const float* I_v       = (const float*)d_in[35];
    const float* I_enc     = (const float*)d_in[36];
    const float* I_attsum  = (const float*)d_in[37];
    const float* I_dec     = (const float*)d_in[38];
    const int*   I_hmask   = (const int*)d_in[39];
    const int*   I_wmask   = (const int*)d_in[40];

    // d_out regions (return order, flat)
    float* O_out   = (float*)d_out;             // 32*128
    float* O_h1    = O_out + BSZ * 128;         // 32*256
    float* O_v1    = O_h1 + BSZ * 256;          // 32*768
    float* O_alpha = O_v1 + BSZ * 768;          // 32*1024
    float* O_atts  = O_alpha + BSZ * SPT;       // 32*1024

    // workspace carve-up
    char* w = (char*)d_ws;
    float* W_emb   = (float*)w;                      w += BSZ * 256 * 4;
    float* W_mask  = (float*)w;                      w += BSZ * SPT * 4;
    float* W_st    = (float*)w;                      w += BSZ * 256 * 4;
    float* W_vt    = (float*)w;                      w += BSZ * 768 * 4;
    float* W_hid1  = (float*)w;                      w += BSZ * 256 * 4;
    float* W_ct    = (float*)w;                      w += BSZ * ENC * 4;
    float* W_pre   = (float*)w;                      w += BSZ * 128 * 4;
    float* W_e     = (float*)w;                      w += BSZ * SPT * 4;
    float* W_esum  = (float*)w;                      w += BSZ * 4;
    float* W_bnsum = (float*)w;                      w += 256 * 4;
    float* W_bnsq  = (float*)w;                      w += 256 * 4;
    float* W_mu    = (float*)w;                      w += 256 * 4;
    float* W_rstd  = (float*)w;                      w += 256 * 4;
    bf16*  W_uawbf = (bf16*)w;                       w += (size_t)ENC * 256 * 2;
    bf16*  W_wtanbf= (bf16*)w;                       w += (size_t)9 * 256 * 256 * 2;
    bf16*  W_et    = (bf16*)w;                       w += (size_t)BSZ * SPT * 256 * 2;
    bf16*  W_xconv = (bf16*)w;                       w += (size_t)BSZ * SPT * 256 * 2;

    k_init<<<1, 256, 0, stream>>>(W_bnsum, W_bnsq, W_esum);
    k_prep<<<BSZ, 256, 0, stream>>>(I_input_a, P_embedding, I_attsum, I_dec,
                                    P_c1_w, P_c1_b, I_hmask, I_wmask,
                                    W_emb, W_mask, O_atts);
    k_cvt<<<(9 * 256 * 256 + 255) / 256, 256, 0, stream>>>(P_ua_w, P_ctan_w, W_uawbf, W_wtanbf);
    // GRU1: x = emb (K=256)
    k_gru<<<(BSZ * HSZ + 255) / 256, 256, 0, stream>>>(W_emb, 256, I_hidden, I_v,
        P_gru1_wi, P_gru1_wh, P_gru1_wv, P_gru1_bi, P_gru1_bh, W_st, W_vt);
    k_hidden1<<<(BSZ * 256 + 255) / 256, 256, 0, stream>>>(W_st, P_hid_w, P_hid_b, W_hid1);
    // dominant GEMM #1: attention projection (WMMA bf16)
    k_ua<<<BSZ * HD, 256, 0, stream>>>(I_enc, W_uawbf, P_ua_b, P_uf_w, P_uf_b,
                                       W_hid1, O_atts, W_et);
    // dominant GEMM #2: implicit-GEMM 3x3 conv (WMMA bf16 + async global->LDS staging)
    k_conv<<<BSZ * HD, 256, 0, stream>>>(W_et, W_wtanbf, P_ctan_b, W_mask,
                                         W_xconv, W_bnsum, W_bnsq);
    k_bnfin<<<1, 256, 0, stream>>>(W_bnsum, W_bnsq, W_mu, W_rstd);
    k_score<<<(BSZ * SPT + 255) / 256, 256, 0, stream>>>(W_xconv, W_mu, W_rstd,
        P_gamma, P_beta, P_vatt_w, P_vatt_b, W_mask, W_e, W_esum);
    k_alpha<<<(BSZ * SPT + 255) / 256, 256, 0, stream>>>(W_e, W_esum, O_alpha);
    k_ct<<<(BSZ * ENC + 255) / 256, 256, 0, stream>>>(O_alpha, I_enc, W_ct);
    // GRU2: x = ct (K=1024)
    k_gru<<<(BSZ * HSZ + 255) / 256, 256, 0, stream>>>(W_ct, 1024, W_st, W_vt,
        P_gru_wi, P_gru_wh, P_gru_wv, P_gru_bi, P_gru_bh, O_h1, O_v1);
    k_prelogit<<<(BSZ * 128 + 255) / 256, 256, 0, stream>>>(O_h1, W_emb, W_ct,
        P_h2_w, P_h2_b, P_e2_w, P_e2_b, P_wc_w, P_wc_b, W_pre);
    k_out<<<BSZ, 128, 0, stream>>>(W_pre, P_out_w, P_out_b, O_out);
}
